// Attention_17763984736760
// MI455X (gfx1250) — matrix operand order repaired and checked by reading, not compiled, not run on hardware
//
#include <hip/hip_runtime.h>

typedef __bf16 bf16;
typedef __bf16 v8bf  __attribute__((ext_vector_type(8)));
typedef __bf16 v16bf __attribute__((ext_vector_type(16)));
typedef float  v8f   __attribute__((ext_vector_type(8)));

#define BB      64
#define NTOK    197
#define DIMC    768
#define NHEAD   12
#define HDIM    64
#define MTOT    (BB * NTOK)   // 12608 = 197 tiles of 64 exactly
#define NPAD    224           // 14 * 16

// ---- fragment loader: two 16B chunks -> one 16-elem bf16 fragment
__device__ __forceinline__ v16bf load_frag(const bf16* p, int o1, int o2) {
  v8bf a = *(const v8bf*)(p + o1);
  v8bf b = *(const v8bf*)(p + o2);
  return __builtin_shufflevector(a, b, 0,1,2,3,4,5,6,7,8,9,10,11,12,13,14,15);
}

__device__ __forceinline__ v8f wmma_bf16(v16bf a, v16bf b, v8f c) {
  return __builtin_amdgcn_wmma_f32_16x16x32_bf16(false, a, false, b, (short)0, c, false, false);
}

// ---- CDNA5 async global->LDS copy (16B per lane), tracked by ASYNCcnt
__device__ __forceinline__ void async_load_b128(const bf16* g, bf16* l) {
  unsigned lds = (unsigned)(uintptr_t)l;           // low 32 bits of generic = LDS byte addr
  unsigned long long ga = (unsigned long long)(uintptr_t)g;
  asm volatile("global_load_async_to_lds_b128 %0, %1, off"
               :: "v"(lds), "v"(ga) : "memory");
}
#define WAIT_ASYNC(n) asm volatile("s_wait_asynccnt %0" :: "n"(n) : "memory")

// ---- fp32 -> bf16 bulk convert (n8 = elems/8)
__global__ void cvt_f32_bf16(const float* __restrict__ s, bf16* __restrict__ d, int n8) {
  int i = blockIdx.x * blockDim.x + threadIdx.x;
  if (i >= n8) return;
  const float4* sp = (const float4*)s;
  float4 a = sp[2*i], b = sp[2*i + 1];
  v8bf o;
  o[0]=(bf16)a.x; o[1]=(bf16)a.y; o[2]=(bf16)a.z; o[3]=(bf16)a.w;
  o[4]=(bf16)b.x; o[5]=(bf16)b.y; o[6]=(bf16)b.z; o[7]=(bf16)b.w;
  *(v8bf*)(d + 8*(size_t)i) = o;
}

// ================= 64x128-tile bf16 GEMM mainloop (double-buffered, async) ==========
// out[m,o] = sum_c A[m,c] * W[o,c], K = 768, BK = 32, 24 stages.
// Wave wv owns a 2x2 grid of 16x16 tiles: rows (wv&1)*32, cols (wv>>1)*32.
// Named accumulators -> in-place WMMA accumulate, no cross-body register copies.
#define GEMM_MAINLOOP(AG, BG)                                                         \
  __shared__ bf16 As[2][64][40];                                                      \
  __shared__ bf16 Bs[2][128][40];                                                     \
  const int tid  = threadIdx.x;                                                       \
  const int lane = tid & 31, wv = tid >> 5;                                           \
  const int hi = lane >> 4, ln = lane & 15;                                           \
  const int tr0 = (wv & 1) * 2;      /* row tile base (of 4)  */                      \
  const int tc0 = (wv >> 1) * 2;     /* col tile base (of 8)  */                      \
  const int lrow = tid >> 2;                                                          \
  const int lcol = (tid & 3) * 8;                                                     \
  v8f acc00 = {}, acc01 = {}, acc10 = {}, acc11 = {};                                 \
  /* stage 0 */                                                                       \
  async_load_b128(AG + (size_t)(m0 + lrow)*DIMC + lcol, &As[0][lrow][lcol]);          \
  async_load_b128(BG + (size_t)(n0 + lrow)*DIMC + lcol, &Bs[0][lrow][lcol]);          \
  async_load_b128(BG + (size_t)(n0 + lrow + 64)*DIMC + lcol, &Bs[0][lrow+64][lcol]);  \
  for (int kt = 0; kt < 24; ++kt) {                                                   \
    const int cur = kt & 1;                                                           \
    if (kt + 1 < 24) {                                                                \
      const int k0 = (kt + 1) * 32;                                                   \
      async_load_b128(AG + (size_t)(m0 + lrow)*DIMC + k0 + lcol,                      \
                      &As[cur ^ 1][lrow][lcol]);                                      \
      async_load_b128(BG + (size_t)(n0 + lrow)*DIMC + k0 + lcol,                      \
                      &Bs[cur ^ 1][lrow][lcol]);                                      \
      async_load_b128(BG + (size_t)(n0 + lrow + 64)*DIMC + k0 + lcol,                 \
                      &Bs[cur ^ 1][lrow+64][lcol]);                                   \
      WAIT_ASYNC(3);                                                                  \
    } else {                                                                          \
      WAIT_ASYNC(0);                                                                  \
    }                                                                                 \
    __syncthreads();                                  /* stage kt visible */          \
    v16bf a0 = load_frag(&As[cur][tr0*16 + ln][0],      8*hi, 16 + 8*hi);             \
    v16bf a1 = load_frag(&As[cur][tr0*16 + 16 + ln][0], 8*hi, 16 + 8*hi);             \
    v16bf b0 = load_frag(&Bs[cur][tc0*16 + ln][0],      8*hi, 16 + 8*hi);             \
    v16bf b1 = load_frag(&Bs[cur][tc0*16 + 16 + ln][0], 8*hi, 16 + 8*hi);             \
    acc00 = wmma_bf16(a0, b0, acc00);                                                 \
    acc01 = wmma_bf16(a0, b1, acc01);                                                 \
    acc10 = wmma_bf16(a1, b0, acc10);                                                 \
    acc11 = wmma_bf16(a1, b1, acc11);                                                 \
    __syncthreads();                                  /* readers done */              \
  }                                                                                   \
  v8f accm[2][2] = {{acc00, acc01}, {acc10, acc11}};  /* one-time for epilogue */

// ---- QKV projection: [12608x768] x [2304x768]^T, scatter into q/k/v [B,H,N,64] bf16
__global__ __launch_bounds__(256) void qkv_gemm(const bf16* __restrict__ xb,
                                                const bf16* __restrict__ wb,
                                                bf16* __restrict__ qb,
                                                bf16* __restrict__ kb,
                                                bf16* __restrict__ vb) {
  const int mt = blockIdx.x % 197;
  const int nt = blockIdx.x / 197;          // 0..17
  const int m0 = mt * 64, n0 = nt * 128;
  GEMM_MAINLOOP(xb, wb)
  for (int i = 0; i < 2; ++i)
    for (int j = 0; j < 2; ++j) {
      int o = n0 + (tc0 + j)*16 + ln;       // 0..2303 = (qkv, head, d)
      int three = o / DIMC;
      int rem = o - three * DIMC;
      int h = rem >> 6, d = rem & 63;
      bf16* dst = (three == 0) ? qb : ((three == 1) ? kb : vb);
      for (int r = 0; r < 8; ++r) {
        int m = m0 + (tr0 + i)*16 + r + 8*hi;
        int b = m / NTOK;
        int n = m - b * NTOK;
        dst[(((size_t)b*NHEAD + h)*NTOK + n)*HDIM + d] = (bf16)accm[i][j][r];
      }
    }
}

// ---- output projection + bias + residual: [12608x768] x [768x768]^T
__global__ __launch_bounds__(256) void proj_gemm(const bf16* __restrict__ ab,
                                                 const bf16* __restrict__ wpb,
                                                 const float* __restrict__ bias,
                                                 const float* __restrict__ x,
                                                 float* __restrict__ out) {
  const int mt = blockIdx.x % 197;
  const int nt = blockIdx.x / 197;          // 0..5
  const int m0 = mt * 64, n0 = nt * 128;
  GEMM_MAINLOOP(ab, wpb)
  for (int i = 0; i < 2; ++i)
    for (int j = 0; j < 2; ++j) {
      int o = n0 + (tc0 + j)*16 + ln;
      float bo = bias[o];
      for (int r = 0; r < 8; ++r) {
        int m = m0 + (tr0 + i)*16 + r + 8*hi;
        size_t idx = (size_t)m * DIMC + o;
        out[idx] = accm[i][j][r] + bo + x[idx];
      }
    }
}

// ---- fused attention per (b,h): S = scale*QK^T, diag/pad mask, softmax, O = P V
__global__ __launch_bounds__(256) void attn_kernel(const bf16* __restrict__ qb,
                                                   const bf16* __restrict__ kb,
                                                   const bf16* __restrict__ vb,
                                                   const float* __restrict__ scale,
                                                   bf16* __restrict__ ob) {
  __shared__ bf16 Ks[NPAD * 64];       // token-major, 28 KB, zero-padded
  __shared__ bf16 Vt[64 * NPAD];       // d-major (transposed), 28 KB, zero-padded
  __shared__ bf16 Ps[8][16 * NPAD];    // per-wave P tiles, 56 KB
  const int bh = blockIdx.x;
  const int b = bh / NHEAD, h = bh - b * NHEAD;
  const bf16* qp = qb + (size_t)bh * NTOK * HDIM;
  const bf16* kp = kb + (size_t)bh * NTOK * HDIM;
  const bf16* vp = vb + (size_t)bh * NTOK * HDIM;
  const int tid = threadIdx.x;
  // K: async global->LDS for in-range tokens (overlaps with V transpose below)
  for (int i = tid * 8; i < NTOK * 64; i += 256 * 8)
    async_load_b128(kp + i, &Ks[i]);
  for (int i = tid * 8; i < NPAD * 64; i += 256 * 8)        // zero pad (disjoint range)
    if (i >= NTOK * 64) *(v8bf*)&Ks[i] = (v8bf){};
  // V: coalesced global read, transposed scatter into Vt[d][token]
  for (int i = tid * 8; i < NTOK * 64; i += 256 * 8) {
    v8bf vz = *(const v8bf*)(vp + i);
    int t = i >> 6, d0 = i & 63;
    for (int e = 0; e < 8; ++e) Vt[(d0 + e) * NPAD + t] = vz[e];
  }
  for (int j = tid; j < 64 * (NPAD - NTOK); j += 256) {     // zero pad tokens 197..223
    int d = j / (NPAD - NTOK), t = NTOK + j % (NPAD - NTOK);
    Vt[d * NPAD + t] = (bf16)0.0f;
  }
  WAIT_ASYNC(0);
  __syncthreads();
  const int lane = tid & 31, wv = tid >> 5;
  const int hi = lane >> 4, ln = lane & 15;
  const float sc = scale[h];
  const float NEG = -__builtin_inff();
  for (int rt = wv; rt < 13; rt += 8) {          // 13 query row-tiles of 16
    int qrow = rt * 16 + ln; if (qrow > NTOK - 1) qrow = NTOK - 1;
    const bf16* qr = qp + (size_t)qrow * HDIM;
    v16bf qf0 = load_frag(qr,      8*hi, 16 + 8*hi);   // d = 0..31
    v16bf qf1 = load_frag(qr + 32, 8*hi, 16 + 8*hi);   // d = 32..63
    v8f s[14];
    for (int j = 0; j < 14; ++j) {               // 14 key col-tiles of 16
      const bf16* kr = &Ks[(j*16 + ln) * 64];
      v16bf kf0 = load_frag(kr,      8*hi, 16 + 8*hi);
      v16bf kf1 = load_frag(kr + 32, 8*hi, 16 + 8*hi);
      v8f a = {};
      a = wmma_bf16(qf0, kf0, a);
      a = wmma_bf16(qf1, kf1, a);
      s[j] = a;
    }
    // scale + diagonal/pad mask
    for (int j = 0; j < 14; ++j) {
      int col = j*16 + ln;
      for (int r = 0; r < 8; ++r) {
        int row = rt*16 + r + 8*hi;
        float v = s[j][r] * sc;
        if (col >= NTOK || col == row) v = NEG;
        s[j][r] = v;
      }
    }
    // row softmax: shuffle-reduce over the 16 lanes of this half-wave
    for (int r = 0; r < 8; ++r) {
      float mx = NEG;
      for (int j = 0; j < 14; ++j) mx = fmaxf(mx, s[j][r]);
      for (int m = 1; m <= 8; m <<= 1) mx = fmaxf(mx, __shfl_xor(mx, m, 32));
      float sum = 0.f;
      for (int j = 0; j < 14; ++j) { float e = __expf(s[j][r] - mx); s[j][r] = e; sum += e; }
      for (int m = 1; m <= 8; m <<= 1) sum += __shfl_xor(sum, m, 32);
      float inv = 1.0f / sum;
      int prow = r + 8*hi;
      for (int j = 0; j < 14; ++j)
        Ps[wv][prow * NPAD + j*16 + ln] = (bf16)(s[j][r] * inv);
    }
    // O = P (16x224) @ V (224x64): 7 K-chunks of 32, 4 col-tiles of 16
    for (int c = 0; c < 4; ++c) {
      v8f o = {};
      for (int kt = 0; kt < 7; ++kt) {
        v16bf pf = load_frag(&Ps[wv][ln * NPAD + kt * 32], 8*hi, 16 + 8*hi);
        v16bf vf = load_frag(&Vt[(c*16 + ln) * NPAD + kt * 32], 8*hi, 16 + 8*hi);
        o = wmma_bf16(pf, vf, o);
      }
      for (int r = 0; r < 8; ++r) {
        int row = rt*16 + r + 8*hi;
        if (row < NTOK)
          ob[((size_t)(b*NTOK + row))*DIMC + h*HDIM + c*16 + ln] = (bf16)o[r];
      }
    }
  }
}

extern "C" void kernel_launch(void* const* d_in, const int* in_sizes, int n_in,
                              void* d_out, int out_size, void* d_ws, size_t ws_size,
                              hipStream_t stream) {
  const float* x      = (const float*)d_in[0];
  const float* scale  = (const float*)d_in[1];
  const float* w_qkv  = (const float*)d_in[2];
  const float* w_proj = (const float*)d_in[3];
  const float* b_proj = (const float*)d_in[4];
  float* out = (float*)d_out;

  char* ws = (char*)d_ws;
  size_t off = 0;
  auto take = [&](size_t elems) -> bf16* {
    bf16* p = (bf16*)(ws + off);
    off = (off + elems * sizeof(bf16) + 255) & ~(size_t)255;
    return p;
  };
  const size_t E_X = (size_t)MTOT * DIMC;     // 9,682,944 elems
  bf16* xb  = take(E_X);                      // x in bf16
  bf16* wqb = take((size_t)3 * DIMC * DIMC);  // w_qkv bf16
  bf16* wpb = take((size_t)DIMC * DIMC);      // w_proj bf16
  bf16* qb  = take(E_X);                      // q [B,H,N,64]
  bf16* kb  = take(E_X);                      // k
  bf16* vb  = take(E_X);                      // v
  bf16* aob = take(E_X);                      // attn out [B,N,768]

  int n8 = (int)(E_X / 8);
  cvt_f32_bf16<<<(n8 + 255) / 256, 256, 0, stream>>>(x, xb, n8);
  n8 = (int)((size_t)3 * DIMC * DIMC / 8);
  cvt_f32_bf16<<<(n8 + 255) / 256, 256, 0, stream>>>(w_qkv, wqb, n8);
  n8 = (int)((size_t)DIMC * DIMC / 8);
  cvt_f32_bf16<<<(n8 + 255) / 256, 256, 0, stream>>>(w_proj, wpb, n8);

  qkv_gemm  <<<197 * 18,   256, 0, stream>>>(xb, wqb, qb, kb, vb);
  attn_kernel<<<BB * NHEAD, 256, 0, stream>>>(qb, kb, vb, scale, aob);
  proj_gemm <<<197 * 6,    256, 0, stream>>>(aob, wpb, b_proj, x, out);
}
